// PaiNNInteraction_34394098106846
// MI455X (gfx1250) — matrix operand order repaired and checked by reading, not compiled, not run on hardware
//
#include <hip/hip_runtime.h>

typedef __attribute__((ext_vector_type(16))) _Float16 v16h;
typedef __attribute__((ext_vector_type(8)))  float    v8f;
typedef __attribute__((ext_vector_type(4)))  float    v4f;
typedef __attribute__((ext_vector_type(2)))  float    v2f;

#define F_DIM 64

// ---------- helpers: build WMMA operands from f32 memory ----------

// A operand (f16 16x32, MxK): lane m = lane&15 is the row; lanes 0-15 hold
// K = kb..kb+7 and kb+16..kb+23 with kb=0, lanes 16-31 with kb=8 (ISA 7.12.2).
__device__ __forceinline__ v16h load_A_f16(const float* __restrict__ row, int ko, int lane) {
  const int kb = (lane >> 4) << 3;              // 0 or 8
  const float* p0 = row + ko + kb;
  const float* p1 = row + ko + 16 + kb;
  v16h a;
#pragma unroll
  for (int i = 0; i < 8; ++i) {
    a[i]     = (_Float16)p0[i];
    a[i + 8] = (_Float16)p1[i];
  }
  return a;
}

// B operand (f16 32x16, KxN) for out = in * W^T with W row-major [G,64]:
// B[k][n] = W[ntile*16+n][k]. Lanes 0-15 hold K=0..7,16..23 of column n, lanes
// 16-31 hold K=8..15,24..31 (matches the A layout).
__device__ __forceinline__ v16h load_B_f16(const float* __restrict__ W, int ntile, int ko, int lane) {
  const int n  = lane & 15;
  const int kb = (lane >> 4) << 3;              // 0 or 8
  const float* p = W + (size_t)(ntile * 16 + n) * F_DIM + ko + kb;
  v16h b;
#pragma unroll
  for (int i = 0; i < 8; ++i) { b[i] = (_Float16)p[i]; b[i + 8] = (_Float16)p[16 + i]; }
  return b;
}

// ---------- kernel 1: x = Dense(192) ( silu( Dense(64)(q) ) ), WMMA ----------

__global__ void __launch_bounds__(256)
painn_mlp_wmma(const float* __restrict__ q, const float* __restrict__ W1,
               const float* __restrict__ b1, const float* __restrict__ W2,
               const float* __restrict__ b2, float* __restrict__ x, int N)
{
  __shared__ _Float16 hsh[8][16 * F_DIM];       // per-wave 16x64 f16 staging
  const int lane = threadIdx.x & 31;
  const int wid  = threadIdx.x >> 5;
  const int tile = blockIdx.x * 8 + wid;        // 16-atom tile per wave
  const int ntiles = (N + 15) >> 4;
  if (tile >= ntiles) return;                   // whole-wave exit: EXEC stays all-1s for WMMA

  const int m  = lane & 15;                     // A row / D column within tile
  const int mh = (lane >> 4) << 3;              // D rows this lane holds: mh..mh+7

  int rowg = tile * 16 + m;
  if (rowg >= N) rowg = N - 1;
  const float* qrow = q + (size_t)rowg * F_DIM;

  v16h a0 = load_A_f16(qrow, 0,  lane);
  v16h a1 = load_A_f16(qrow, 32, lane);

  // GEMM1: h[16,64] = silu(q[16,64] @ W1^T + b1)  -> LDS (f16)
#pragma unroll
  for (int nt = 0; nt < 4; ++nt) {
    v8f c = {};
    v16h bb0 = load_B_f16(W1, nt, 0,  lane);
    v16h bb1 = load_B_f16(W1, nt, 32, lane);
    c = __builtin_amdgcn_wmma_f32_16x16x32_f16(false, a0, false, bb0, (short)0, c, false, false);
    c = __builtin_amdgcn_wmma_f32_16x16x32_f16(false, a1, false, bb1, (short)0, c, false, false);
    const float bias = b1[nt * 16 + m];
#pragma unroll
    for (int r = 0; r < 8; ++r) {
      float v = c[r] + bias;
      v = v * (1.0f / (1.0f + __expf(-v)));     // SiLU
      hsh[wid][(mh + r) * F_DIM + nt * 16 + m] = (_Float16)v;
    }
  }

  // LDS store->load turnaround within this wave
  asm volatile("s_wait_dscnt 0" ::: "memory");

  // reload h in A-operand layout
  const _Float16* hrow = &hsh[wid][m * F_DIM];
  const int kb = (lane >> 4) << 3;
  v16h h0, h1;
#pragma unroll
  for (int i = 0; i < 8; ++i) {
    h0[i]     = hrow[kb + i];
    h0[i + 8] = hrow[16 + kb + i];
    h1[i]     = hrow[32 + kb + i];
    h1[i + 8] = hrow[48 + kb + i];
  }

  // GEMM2: x[16,192] = h @ W2^T + b2
#pragma unroll
  for (int nt = 0; nt < 12; ++nt) {
    v8f c = {};
    v16h bb0 = load_B_f16(W2, nt, 0,  lane);
    v16h bb1 = load_B_f16(W2, nt, 32, lane);
    c = __builtin_amdgcn_wmma_f32_16x16x32_f16(false, h0, false, bb0, (short)0, c, false, false);
    c = __builtin_amdgcn_wmma_f32_16x16x32_f16(false, h1, false, bb1, (short)0, c, false, false);
    const float bias = b2[nt * 16 + m];
#pragma unroll
    for (int r = 0; r < 8; ++r) {
      const int row = tile * 16 + mh + r;
      if (row < N) x[(size_t)row * 192 + nt * 16 + m] = c[r] + bias;
    }
  }
}

// ---------- kernel 2: out = [q, mu] (atomics accumulate on top) ----------

__global__ void __launch_bounds__(256)
painn_init_out(const float* __restrict__ q, const float* __restrict__ mu,
               float* __restrict__ out, long nq, long total)
{
  long i4 = (long)blockIdx.x * blockDim.x + threadIdx.x;
  long base = i4 << 2;
  if (base >= total) return;
  v4f v;
  if (base < nq) v = *(const v4f*)(q + base);
  else           v = *(const v4f*)(mu + (base - nq));
  *(v4f*)(out + base) = v;
}

// ---------- kernel 3: per-edge gather / scale / scatter-add ----------

__global__ void __launch_bounds__(256)
painn_edge(const float* __restrict__ Wij, const float* __restrict__ x,
           const float* __restrict__ mu, const float* __restrict__ dir_ij,
           const int* __restrict__ idx_i, const int* __restrict__ idx_j,
           float* __restrict__ outq, float* __restrict__ outmu, int E)
{
  const int e    = (int)(((long)blockIdx.x * blockDim.x + threadIdx.x) >> 5); // one wave32 per edge
  const int lane = threadIdx.x & 31;
  if (e >= E) return;

  const int i = idx_i[e];
  const int j = idx_j[e];
  const int f = lane << 1;                      // 2 features per lane (F=64)

  // Wij: read-once 614MB stream -> non-temporal, keep L2 for x/mu tables
  const float* we = Wij + (size_t)e * 192;
  const v2f w_q = __builtin_nontemporal_load((const v2f*)(we + f));
  const v2f w_r = __builtin_nontemporal_load((const v2f*)(we + 64 + f));
  const v2f w_m = __builtin_nontemporal_load((const v2f*)(we + 128 + f));

  // gathers: x (38MB) and mu (38MB) are L2-resident
  const float* xj = x + (size_t)j * 192;
  const v2f x_q = *(const v2f*)(xj + f);
  const v2f x_r = *(const v2f*)(xj + 64 + f);
  const v2f x_m = *(const v2f*)(xj + 128 + f);

  const v2f dq    = w_q * x_q;
  const v2f dmuR  = w_r * x_r;
  const v2f dmumu = w_m * x_m;

  const float d0 = dir_ij[(size_t)e * 3 + 0];
  const float d1 = dir_ij[(size_t)e * 3 + 1];
  const float d2 = dir_ij[(size_t)e * 3 + 2];

  const float* muj = mu + (size_t)j * 192;
  const v2f m0 = *(const v2f*)(muj + f);
  const v2f m1 = *(const v2f*)(muj + 64 + f);
  const v2f m2 = *(const v2f*)(muj + 128 + f);

  const v2f u0 = dmuR * d0 + dmumu * m0;
  const v2f u1 = dmuR * d1 + dmumu * m1;
  const v2f u2 = dmuR * d2 + dmumu * m2;

  // segment_sum -> L2 fp32 atomics (global_atomic_add_f32)
  float* oq = outq + (size_t)i * 64 + f;
  unsafeAtomicAdd(oq + 0, dq.x);
  unsafeAtomicAdd(oq + 1, dq.y);
  float* om = outmu + (size_t)i * 192 + f;
  unsafeAtomicAdd(om + 0,   u0.x);
  unsafeAtomicAdd(om + 1,   u0.y);
  unsafeAtomicAdd(om + 64,  u1.x);
  unsafeAtomicAdd(om + 65,  u1.y);
  unsafeAtomicAdd(om + 128, u2.x);
  unsafeAtomicAdd(om + 129, u2.y);
}

// ---------- launch ----------

extern "C" void kernel_launch(void* const* d_in, const int* in_sizes, int n_in,
                              void* d_out, int out_size, void* d_ws, size_t ws_size,
                              hipStream_t stream) {
  (void)n_in; (void)out_size; (void)ws_size;
  const float* q   = (const float*)d_in[0];
  const float* mu  = (const float*)d_in[1];
  const float* Wij = (const float*)d_in[2];
  const float* dir = (const float*)d_in[3];
  const int* idx_i = (const int*)d_in[4];
  const int* idx_j = (const int*)d_in[5];
  // d_in[6] = n_atoms scalar (derive N from sizes instead)
  const float* W1  = (const float*)d_in[7];
  const float* b1  = (const float*)d_in[8];
  const float* W2  = (const float*)d_in[9];
  const float* b2  = (const float*)d_in[10];

  const int N = in_sizes[0] / F_DIM;
  const int E = in_sizes[4];

  float* outq  = (float*)d_out;
  float* outmu = outq + (size_t)N * F_DIM;
  float* xws   = (float*)d_ws;                  // x[N,192] f32, L2-resident gather table

  {   // out = [q, mu]
    const long nq = (long)N * 64;
    const long total = nq + (long)N * 192;
    const int blocks = (int)((total / 4 + 255) / 256);
    painn_init_out<<<blocks, 256, 0, stream>>>(q, mu, (float*)d_out, nq, total);
  }
  {   // x = MLP(q) via WMMA
    const int ntiles = (N + 15) / 16;
    const int blocks = (ntiles + 7) / 8;        // 8 waves (tiles) per block
    painn_mlp_wmma<<<blocks, 256, 0, stream>>>(q, W1, b1, W2, b2, xws, N);
  }
  {   // edge message + scatter
    const int blocks = (E + 7) / 8;             // 8 edges (waves) per block
    painn_edge<<<blocks, 256, 0, stream>>>(Wij, xws, mu, dir, idx_i, idx_j, outq, outmu, E);
  }
}
